// Generator_27539330302298
// MI455X (gfx1250) — matrix operand order repaired and checked by reading, not compiled, not run on hardware
//
#include <hip/hip_runtime.h>
#include <hip/hip_bf16.h>

// ---- CDNA5 WMMA types -------------------------------------------------------
typedef __attribute__((ext_vector_type(16))) _Float16 v16h;
typedef __attribute__((ext_vector_type(8)))  float    v8f;

#define HIDDEN 64
#define NOISE_F 32
#define NPL 8
#define RES 1024
#define KEXT 96            // 64 activations + 8 noise + 1 bias + 23 zero pad
#define WAVES_PER_BLK 8
#define THREADS (WAVES_PER_BLK * 32)
#define PIX_PER_BLK (WAVES_PER_BLK * 16)   // 128 pixels / block

static __device__ inline v16h ld16h(const _Float16* p0, const _Float16* p1) {
    // Two 16-byte LDS chunks -> one 16-half WMMA fragment register group
    union { v16h v; int4 q[2]; } u;
    u.q[0] = *reinterpret_cast<const int4*>(p0);
    u.q[1] = *reinterpret_cast<const int4*>(p1);
    return u.v;
}

static __device__ inline v8f wmma16(v16h a, v16h b, v8f c) {
    // D(16x16,f32) = A(16x32,f16) x B(32x16,f16) + C
    return __builtin_amdgcn_wmma_f32_16x16x32_f16(
        /*neg_a=*/false, a, /*neg_b=*/false, b,
        /*c_mod=*/(short)0, c, /*reuse_a=*/false, /*reuse_b=*/false);
}

__global__ __launch_bounds__(THREADS)
void Generator_27539330302298_kernel(
    const float* __restrict__ coords,     // [1024*1024*2]
    const float* __restrict__ image,      // [32*1024*1024]
    const float* __restrict__ transforms, // [32*2*2]
    const float* __restrict__ inp,        // [64]
    const float* __restrict__ w_lin,      // [4*64*64]
    const float* __restrict__ b_lin,      // [4*64]
    const float* __restrict__ w_noise,    // [4*64*8]
    const float* __restrict__ tail_w1,    // [64*64]
    const float* __restrict__ tail_b1,    // [64]
    const float* __restrict__ tail_w2,    // [64]
    const float* __restrict__ tail_b2,    // [1]
    float* __restrict__ out)              // [1024*1024]
{
    // LDS staging (static, ~45 KB total; WGP has 320 KB)
    __shared__ __align__(16) _Float16 sW[HIDDEN * KEXT];        // 12 KB, one layer's B matrix (ext)
    __shared__ __align__(16) _Float16 sX[PIX_PER_BLK * KEXT];   // 24 KB, activations (ext A matrix)
    __shared__ __align__(16) _Float16 sNV[PIX_PER_BLK * NOISE_F]; // 8 KB, per-pixel noise samples
    __shared__ float sW2[HIDDEN];                               // 256 B

    const int tid  = threadIdx.x;
    const int lane = tid & 31;
    const int wv   = tid >> 5;         // wave id in block
    const int hi   = lane >> 4;        // half-wave selector (K split)
    const int lo   = lane & 15;        // row (A) / col (B,C) within tile
    const long blockBase = (long)blockIdx.x * PIX_PER_BLK;

    // ---- stage final projection vector ----
    if (tid < HIDDEN) sW2[tid] = tail_w2[tid];

    // ---- init activation matrix: X = inp (f16), bias column = 1, pad = 0 ----
    for (int idx = tid; idx < PIX_PER_BLK * KEXT; idx += THREADS) {
        const int k = idx % KEXT;
        _Float16 v;
        if (k < HIDDEN)      v = (_Float16)inp[k];
        else if (k == 72)    v = (_Float16)1.0f;
        else                 v = (_Float16)0.0f;
        sX[idx] = v;
    }

    // ---- per-pixel noise sampling: lane handles 1 pixel x 16 features ----
    {
        const int  pr  = wv * 16 + lo;            // pixel row within block
        const long pix = blockBase + pr;
        const float cx = coords[2 * pix + 0];
        const float cy = coords[2 * pix + 1];
        #pragma unroll 4
        for (int t = 0; t < 16; ++t) {
            const int f = hi * 16 + t;            // noise feature
            const float* T = transforms + f * 4;  // [m][n] row-major 2x2
            const float nx = cx * T[0] + cy * T[1];
            const float ny = cx * T[2] + cy * T[3];
            const float x  = nx - 0.5f;
            const float y  = ny - 0.5f;
            const float fx = floorf(x), fy = floorf(y);
            const float xw = x - fx,   yw = y - fy;
            const int ix0 = ((int)fx) & (RES - 1);
            const int iy0 = ((int)fy) & (RES - 1);
            const int ix1 = (ix0 + 1) & (RES - 1);
            const int iy1 = (iy0 + 1) & (RES - 1);
            const float* img = image + (size_t)f * RES * RES;
            const float i00 = img[iy0 * RES + ix0];
            const float i01 = img[iy0 * RES + ix1];
            const float i10 = img[iy1 * RES + ix0];
            const float i11 = img[iy1 * RES + ix1];
            const float i0 = i00 + xw * (i01 - i00);
            const float i1 = i10 + xw * (i11 - i10);
            sNV[pr * NOISE_F + f] = (_Float16)(i0 + yw * (i1 - i0));
        }
    }
    __syncthreads();

    // ---- 5 layers: 4 noise layers + tail1, all as one extended GEMM form ----
    for (int L = 0; L < 5; ++L) {
        // Stage B_ext[k][n] = Wext[n][k]: row n = out unit, 96 K entries
        for (int idx = tid; idx < HIDDEN * KEXT; idx += THREADS) {
            const int h = idx / KEXT;
            const int k = idx % KEXT;
            float v = 0.0f;
            if (L < 4) {
                if (k < HIDDEN)      v = w_lin[(size_t)L * HIDDEN * HIDDEN + h * HIDDEN + k];
                else if (k < 72)     v = w_noise[(size_t)L * HIDDEN * NPL + h * NPL + (k - 64)];
                else if (k == 72)    v = b_lin[L * HIDDEN + h];
            } else {
                if (k < HIDDEN)      v = tail_w1[h * HIDDEN + k];
                else if (k == 72)    v = tail_b1[h];
            }
            sW[idx] = (_Float16)v;
        }
        // Refresh noise columns (64..71) of A_ext for this layer
        if (tid < PIX_PER_BLK) {
            int4* dst = reinterpret_cast<int4*>(&sX[tid * KEXT + 64]);
            if (L < 4) {
                *dst = *reinterpret_cast<const int4*>(&sNV[tid * NOISE_F + L * NPL]);
            } else {
                int4 z; z.x = z.y = z.z = z.w = 0;
                *dst = z;
            }
        }
        __syncthreads();

        // Y[16x64] = A_ext[16x96] x B_ext[96x64]; 4 N-tiles x 3 K-steps
        v8f acc0 = {}, acc1 = {}, acc2 = {}, acc3 = {};
        #pragma unroll
        for (int ks = 0; ks < 3; ++ks) {
            const int kk = ks * 32;
            // A fragment: lane lo = row, two contiguous 8-half chunks
            const _Float16* arow = &sX[(wv * 16 + lo) * KEXT + kk + hi * 16];
            const v16h a = ld16h(arow, arow + 8);
            // B fragments: lane lo = out unit within tile, 16 contiguous halfs
            const _Float16* b0p = &sW[(0 * 16 + lo) * KEXT + kk + hi * 16];
            const _Float16* b1p = &sW[(1 * 16 + lo) * KEXT + kk + hi * 16];
            const _Float16* b2p = &sW[(2 * 16 + lo) * KEXT + kk + hi * 16];
            const _Float16* b3p = &sW[(3 * 16 + lo) * KEXT + kk + hi * 16];
            acc0 = wmma16(a, ld16h(b0p, b0p + 8), acc0);
            acc1 = wmma16(a, ld16h(b1p, b1p + 8), acc1);
            acc2 = wmma16(a, ld16h(b2p, b2p + 8), acc2);
            acc3 = wmma16(a, ld16h(b3p, b3p + 8), acc3);
        }

        // Leaky-ReLU + scatter C back into A layout (lane holds 1 unit x 8 pixels)
        #pragma unroll
        for (int v = 0; v < 8; ++v) {
            const int m = v + hi * 8;              // pixel row
            float y0 = acc0[v]; y0 = y0 > 0.0f ? y0 : 0.01f * y0;
            float y1 = acc1[v]; y1 = y1 > 0.0f ? y1 : 0.01f * y1;
            float y2 = acc2[v]; y2 = y2 > 0.0f ? y2 : 0.01f * y2;
            float y3 = acc3[v]; y3 = y3 > 0.0f ? y3 : 0.01f * y3;
            _Float16* xr = &sX[(wv * 16 + m) * KEXT];
            xr[0 * 16 + lo] = (_Float16)y0;
            xr[1 * 16 + lo] = (_Float16)y1;
            xr[2 * 16 + lo] = (_Float16)y2;
            xr[3 * 16 + lo] = (_Float16)y3;
        }
        __syncthreads();
    }

    // ---- final 64 -> 1 projection ----
    if (tid < PIX_PER_BLK) {
        float sum = tail_b2[0];
        const _Float16* hrow = &sX[tid * KEXT];
        #pragma unroll
        for (int h = 0; h < HIDDEN; ++h) sum += (float)hrow[h] * sW2[h];
        out[blockBase + tid] = sum;
    }
}

extern "C" void kernel_launch(void* const* d_in, const int* in_sizes, int n_in,
                              void* d_out, int out_size, void* d_ws, size_t ws_size,
                              hipStream_t stream) {
    const float* coords     = (const float*)d_in[0];
    const float* image      = (const float*)d_in[1];
    const float* transforms = (const float*)d_in[2];
    const float* inp        = (const float*)d_in[3];
    const float* w_lin      = (const float*)d_in[4];
    const float* b_lin      = (const float*)d_in[5];
    const float* w_noise    = (const float*)d_in[6];
    const float* tail_w1    = (const float*)d_in[7];
    const float* tail_b1    = (const float*)d_in[8];
    const float* tail_w2    = (const float*)d_in[9];
    const float* tail_b2    = (const float*)d_in[10];
    float* out = (float*)d_out;

    const int totalPix = 1024 * 1024;
    const int blocks = totalPix / PIX_PER_BLK;   // 8192
    Generator_27539330302298_kernel<<<blocks, THREADS, 0, stream>>>(
        coords, image, transforms, inp, w_lin, b_lin, w_noise,
        tail_w1, tail_b1, tail_w2, tail_b2, out);
}
